// WLAMBlock_15522011808258
// MI455X (gfx1250) — matrix operand order repaired and checked by reading, not compiled, not run on hardware
//
#include <hip/hip_runtime.h>
#include <math.h>

typedef __attribute__((ext_vector_type(16))) __bf16 v16bf;
typedef __attribute__((ext_vector_type(8)))  float  v8f;

#define C_DIM 1024
#define C4    256
#define HEADS 16
#define HD 64
#define B_DIM 8
#define S_DIM 4096
#define N_LOW 1024
#define KVZ_STRIDE (HD * HD + HD)
#define NSEG 8

union FragU { uint4 u[2]; v16bf v; };

__device__ __forceinline__ unsigned int pack_bf16(float a, float b) {
    union { __bf16 h; unsigned short s; } x, y;
    x.h = (__bf16)a; y.h = (__bf16)b;
    return (unsigned int)x.s | ((unsigned int)y.s << 16);
}

__device__ __forceinline__ float elu1(float x) {
    // elu(x)+1 with alpha=1:  x>0 ? x+1 : exp(x)
    return x > 0.0f ? x + 1.0f : __expf(x);
}

__device__ __forceinline__ float4 fma4(float4 a, float4 b, float4 c) {
    float4 r;
    r.x = fmaf(a.x, b.x, c.x); r.y = fmaf(a.y, b.y, c.y);
    r.z = fmaf(a.z, b.z, c.z); r.w = fmaf(a.w, b.w, c.w);
    return r;
}

// ---------------------------------------------------------------------------
// DWT analysis: depthwise K=4 'SAME' conv (pad_lo=1) + stride-2 decimation.
// One thread per (b, m, c4) float4 of channels; computes low AND high band.
// ---------------------------------------------------------------------------
__global__ void k_dwt_analysis(const float* __restrict__ x,
                               const float* __restrict__ hf,
                               const float* __restrict__ gf,
                               float* __restrict__ low,
                               float* __restrict__ high,
                               int s_in, int s_half, int total4)
{
    int idx = blockIdx.x * blockDim.x + threadIdx.x;
    if (idx >= total4) return;
    int c4   = (idx & (C4 - 1)) * 4;
    int rest = idx >> 8;
    int m    = rest % s_half;
    int b    = rest / s_half;
    const float* xb = x + ((size_t)b * s_in) * C_DIM + c4;
    float4 al = {0.f, 0.f, 0.f, 0.f};
    float4 ah = al;
#pragma unroll
    for (int k = 0; k < 4; ++k) {
        int n = 2 * m - 1 + k;
        if (n >= 0 && n < s_in) {
            float4 v = *(const float4*)&xb[(size_t)n * C_DIM];
            al = fma4(v, *(const float4*)&hf[k * C_DIM + c4], al);
            ah = fma4(v, *(const float4*)&gf[k * C_DIM + c4], ah);
        }
    }
    *(float4*)&low[(size_t)idx * 4]  = al;
    *(float4*)&high[(size_t)idx * 4] = ah;
}

// ---------------------------------------------------------------------------
// Linear attention pass A: kv[d,e] = sum_n q[n,d]*v[n,e],  z[d] = sum_n q[n,d]
// Grid: B*H*NSEG workgroups of 256 (8 waves). q/v are staged directly into
// WMMA fragment layout in LDS (A-frag / B-frag major), so each wave reads its
// fragments as two ds_load_b128 per matrix. z accumulated in registers (fixed
// channel per thread) and flushed with one global atomic per thread.
// ---------------------------------------------------------------------------
__global__ __launch_bounds__(256) void k_attn_kv(const float* __restrict__ low2,
                                                 float* __restrict__ kvz)
{
    // Fragment-major staging: [tile][lane][slot]
    __shared__ __align__(16) __bf16 qA[4][32][16];   // A-frag: q (d-dim tiles)
    __shared__ __align__(16) __bf16 vB[4][32][16];   // B-frag: v (e-dim tiles)

    int seg  = blockIdx.x % NSEG;
    int bh   = blockIdx.x / NSEG;
    int b    = bh / HEADS;
    int h    = bh % HEADS;
    int tid  = threadIdx.x;
    int lane = tid & 31;
    int wave = tid >> 5;
    int laneM = lane & 15;
    int hi    = lane >> 4;

    v8f acc0 = {};
    v8f acc1 = {};
    float zpart = 0.f;                 // partial z for channel d = tid & 63

    const float* base = low2 + ((size_t)b * N_LOW) * C_DIM + h * HD;
    int n_start = seg * (N_LOW / NSEG);

    for (int chunk = 0; chunk < (N_LOW / NSEG) / 32; ++chunk) {
        int n0 = n_start + chunk * 32;
        // stage 32 rows x 64 channels; pairs over kk so stores are b32-packed
#pragma unroll
        for (int i = 0; i < 4; ++i) {
            int p  = tid + 256 * i;            // pair index: 16 kk-pairs x 64 d
            int d  = p & 63;                   // == tid & 63 (fixed per thread)
            int kk = (p >> 6) * 2;
            float v0 = base[(size_t)(n0 + kk) * C_DIM + d];
            float v1 = base[(size_t)(n0 + kk + 1) * C_DIM + d];
            float q0 = elu1(v0), q1 = elu1(v1);
            zpart += q0 + q1;
            // A fragment scatter: lane-half from (kk>>3)&1, slot=(kk&7)+8*((kk>>4)&1)
            int laneA = (d & 15) + 16 * ((kk >> 3) & 1);
            int slotA = (kk & 7) + 8 * ((kk >> 4) & 1);
            *(unsigned int*)&qA[d >> 4][laneA][slotA] = pack_bf16(q0, q1);
            // B fragment scatter: lane-half from (kk>>3)&1, slot=(kk&7)+8*((kk>>4)&1)
            int laneB = (d & 15) + 16 * ((kk >> 3) & 1);
            int slotB = (kk & 7) + 8 * ((kk >> 4) & 1);
            *(unsigned int*)&vB[d >> 4][laneB][slotB] = pack_bf16(v0, v1);
        }
        __syncthreads();

#pragma unroll
        for (int i = 0; i < 2; ++i) {
            int tile = 2 * wave + i;
            int et = tile >> 2;     // kv row (d-dim) tile
            int ft = tile & 3;      // kv col (e-dim) tile
            FragU fa, fb;
            const uint4* pa = (const uint4*)&qA[et][lane][0];
            const uint4* pb = (const uint4*)&vB[ft][lane][0];
            fa.u[0] = pa[0]; fa.u[1] = pa[1];
            fb.u[0] = pb[0]; fb.u[1] = pb[1];
            v8f acc = i ? acc1 : acc0;
            acc = __builtin_amdgcn_wmma_f32_16x16x32_bf16(
                false, fa.v, false, fb.v, (short)0, acc, false, false);
            if (i) acc1 = acc; else acc0 = acc;
        }
        __syncthreads();
    }

    float* kv = kvz + (size_t)bh * KVZ_STRIDE;
#pragma unroll
    for (int i = 0; i < 2; ++i) {
        int tile = 2 * wave + i;
        int e0 = (tile >> 2) * 16;
        int f0 = (tile & 3) * 16;
        v8f acc = i ? acc1 : acc0;
#pragma unroll
        for (int r = 0; r < 8; ++r) {
            int M = r + (hi ? 8 : 0);
            atomicAdd(&kv[(e0 + M) * HD + f0 + laneM], acc[r]);
        }
    }
    atomicAdd(&kv[HD * HD + (tid & 63)], zpart);
}

// ---------------------------------------------------------------------------
// Linear attention pass B: out[n,e] = (sum_d q[n,d]*kv[d,e]) / (q[n,:].z + eps)
// q rows staged row-major (A-frag = two contiguous uint4 runs per lane);
// kv restaged into B-fragment layout. Writes in place into low2.
// ---------------------------------------------------------------------------
__global__ __launch_bounds__(256) void k_attn_apply(float* __restrict__ att,
                                                    const float* __restrict__ kvz)
{
    __shared__ __align__(16) __bf16 qs[64][72];        // row-major q (padded)
    __shared__ __align__(16) __bf16 kvB[4][2][32][16]; // B-frag kv [e-tile][dchunk]
    __shared__ float  zsh[HD];
    __shared__ float  den[64];

    int nblk = blockIdx.x & 15;
    int bh   = blockIdx.x >> 4;
    int b    = bh / HEADS;
    int h    = bh % HEADS;
    int tid  = threadIdx.x;
    int lane = tid & 31;
    int wave = tid >> 5;
    int laneN = lane & 15;
    int hi    = lane >> 4;
    int n0 = nblk * 64;

    const float* kvp = kvz + (size_t)bh * KVZ_STRIDE;
    float* base = att + (((size_t)b * N_LOW) + n0) * C_DIM + h * HD;

    // stage kv into B-fragment layout (pairs over d -> packed b32 stores)
#pragma unroll
    for (int i = 0; i < 8; ++i) {
        int p  = tid + 256 * i;          // 2048 pairs: 32 d-pairs x 64 e
        int e  = p & 63;
        int d  = (p >> 6) * 2;
        float k0 = kvp[d * HD + e];
        float k1 = kvp[(d + 1) * HD + e];
        int kk = d & 31;
        int laneB = (e & 15) + 16 * ((kk >> 3) & 1);
        *(unsigned int*)&kvB[e >> 4][d >> 5][laneB][(kk & 7) + 8 * ((kk >> 4) & 1)] = pack_bf16(k0, k1);
    }
    if (tid < HD) zsh[tid] = kvp[HD * HD + tid];
    // stage q rows (pairs over d -> float2 loads, packed b32 stores)
#pragma unroll
    for (int i = 0; i < 8; ++i) {
        int p = tid + 256 * i;           // 2048 pairs: 64 rows x 32 d-pairs
        int r = p >> 5;
        int d = (p & 31) * 2;
        float2 v = *(const float2*)&base[(size_t)r * C_DIM + d];
        *(unsigned int*)&qs[r][d] = pack_bf16(elu1(v.x), elu1(v.y));
    }
    __syncthreads();

    if (tid < 64) {
        float s = 1e-6f;
#pragma unroll
        for (int d = 0; d < HD; ++d) s += (float)qs[tid][d] * zsh[d];
        den[tid] = s;
    }
    __syncthreads();

#pragma unroll
    for (int i = 0; i < 2; ++i) {
        int tile = 2 * wave + i;
        int m0 = (tile >> 2) * 16;   // query-row tile offset within 64
        int et = tile & 3;           // e-dim tile
        int e0 = et * 16;
        v8f acc = {};
#pragma unroll
        for (int dc = 0; dc < 2; ++dc) {   // K = 64 in two chunks of 32
            int d0 = dc * 32;
            FragU fa, fb;
            // A frag: row m0+laneN, K segments {0..7,16..23} (+8 for hi half)
            const uint4* ra = (const uint4*)&qs[m0 + laneN][d0];
            fa.u[0] = ra[hi];         // hi=0: K 0..7  | hi=1: K 8..15
            fa.u[1] = ra[2 + hi];     // hi=0: K 16..23| hi=1: K 24..31
            const uint4* pb = (const uint4*)&kvB[et][dc][lane][0];
            fb.u[0] = pb[0]; fb.u[1] = pb[1];
            acc = __builtin_amdgcn_wmma_f32_16x16x32_bf16(
                false, fa.v, false, fb.v, (short)0, acc, false, false);
        }
#pragma unroll
        for (int r = 0; r < 8; ++r) {
            int M = r + (hi ? 8 : 0);
            base[(size_t)(m0 + M) * C_DIM + e0 + laneN] = acc[r] / den[m0 + M];
        }
    }
}

// ---------------------------------------------------------------------------
// IWT synthesis (deepest level): repeat-x2 upsample folded into m>>1 index,
// K=4 'SAME' conv with synthesis filters, sum of both bands. float4 channels.
// ---------------------------------------------------------------------------
__global__ void k_synth(const float* __restrict__ lowu,
                        const float* __restrict__ highu,
                        const float* __restrict__ hs,
                        const float* __restrict__ gs,
                        float* __restrict__ out,
                        int s_half, int total4)
{
    int idx = blockIdx.x * blockDim.x + threadIdx.x;
    if (idx >= total4) return;
    int c4   = (idx & (C4 - 1)) * 4;
    int rest = idx >> 8;
    int s_out = 2 * s_half;
    int n = rest % s_out;
    int b = rest / s_out;
    const float* lb = lowu  + ((size_t)b * s_half) * C_DIM + c4;
    const float* hb = highu + ((size_t)b * s_half) * C_DIM + c4;
    float4 acc = {0.f, 0.f, 0.f, 0.f};
#pragma unroll
    for (int k = 0; k < 4; ++k) {
        int m = n - 1 + k;
        if (m >= 0 && m < s_out) {
            int src = m >> 1;
            acc = fma4(*(const float4*)&lb[(size_t)src * C_DIM],
                       *(const float4*)&hs[k * C_DIM + c4], acc);
            acc = fma4(*(const float4*)&hb[(size_t)src * C_DIM],
                       *(const float4*)&gs[k * C_DIM + c4], acc);
        }
    }
    *(float4*)&out[(size_t)idx * 4] = acc;
}

// ---------------------------------------------------------------------------
// Fused: IWT synthesis level 1 + residual + LayerNorm.
// One block per (b, n) row; one float4 of channels per thread; LDS reduction.
// ---------------------------------------------------------------------------
__global__ __launch_bounds__(256) void k_synth_ln(const float* __restrict__ rec1,
                                                  const float* __restrict__ high1,
                                                  const float* __restrict__ x,
                                                  const float* __restrict__ hs,
                                                  const float* __restrict__ gs,
                                                  const float* __restrict__ gamma,
                                                  const float* __restrict__ beta,
                                                  float* __restrict__ out)
{
    __shared__ float rs[256], rs2[256];
    int row = blockIdx.x;            // b*4096 + n
    int b   = row >> 12;
    int n   = row & (S_DIM - 1);
    int tid = threadIdx.x;
    int c4  = tid * 4;
    const float* lb = rec1  + ((size_t)b * (S_DIM / 2)) * C_DIM + c4;
    const float* hb = high1 + ((size_t)b * (S_DIM / 2)) * C_DIM + c4;

    float4 acc = {0.f, 0.f, 0.f, 0.f};
#pragma unroll
    for (int k = 0; k < 4; ++k) {
        int m = n - 1 + k;
        if (m >= 0 && m < S_DIM) {
            int src = m >> 1;
            acc = fma4(*(const float4*)&lb[(size_t)src * C_DIM],
                       *(const float4*)&hs[k * C_DIM + c4], acc);
            acc = fma4(*(const float4*)&hb[(size_t)src * C_DIM],
                       *(const float4*)&gs[k * C_DIM + c4], acc);
        }
    }
    float4 xv = *(const float4*)&x[(size_t)row * C_DIM + c4];
    float4 y;
    y.x = xv.x + acc.x; y.y = xv.y + acc.y;
    y.z = xv.z + acc.z; y.w = xv.w + acc.w;

    float s  = y.x + y.y + y.z + y.w;
    float s2 = y.x * y.x + y.y * y.y + y.z * y.z + y.w * y.w;
    rs[tid] = s; rs2[tid] = s2;
    __syncthreads();
    for (int off = 128; off > 0; off >>= 1) {
        if (tid < off) { rs[tid] += rs[tid + off]; rs2[tid] += rs2[tid + off]; }
        __syncthreads();
    }
    float mu  = rs[0] * (1.0f / C_DIM);
    float var = rs2[0] * (1.0f / C_DIM) - mu * mu;
    float inv = rsqrtf(var + 1e-5f);
    float4 g = *(const float4*)&gamma[c4];
    float4 be = *(const float4*)&beta[c4];
    float4 o;
    o.x = (y.x - mu) * inv * g.x + be.x;
    o.y = (y.y - mu) * inv * g.y + be.y;
    o.z = (y.z - mu) * inv * g.z + be.z;
    o.w = (y.w - mu) * inv * g.w + be.w;
    *(float4*)&out[(size_t)row * C_DIM + c4] = o;
}

// ---------------------------------------------------------------------------
extern "C" void kernel_launch(void* const* d_in, const int* in_sizes, int n_in,
                              void* d_out, int out_size, void* d_ws, size_t ws_size,
                              hipStream_t stream)
{
    const float* x        = (const float*)d_in[0];
    const float* h_filter = (const float*)d_in[1];
    const float* g_filter = (const float*)d_in[2];
    const float* h_synth  = (const float*)d_in[3];
    const float* g_synth  = (const float*)d_in[4];
    const float* gamma    = (const float*)d_in[5];
    const float* beta     = (const float*)d_in[6];
    float* out = (float*)d_out;

    float* ws = (float*)d_ws;
    const size_t L1  = (size_t)B_DIM * (S_DIM / 2) * C_DIM;   // 16.8M floats
    const size_t L2  = (size_t)B_DIM * (S_DIM / 4) * C_DIM;   //  8.4M floats
    float* low1  = ws;                 // [B,2048,C]
    float* high1 = ws + L1;            // [B,2048,C]
    float* low2  = ws + 2 * L1;        // [B,1024,C]  (att written in place)
    float* high2 = ws + 2 * L1 + L2;   // [B,1024,C]
    float* kvz   = ws + 2 * L1 + 2 * L2;               // [B*H, 64*64+64]
    float* rec1  = low1;               // reuse: low1 dead after level-2 analysis

    // DWT analysis level 1:  x -> low1, high1
    {
        int total4 = (int)(L1 / 4);
        k_dwt_analysis<<<total4 / 256, 256, 0, stream>>>(
            x, h_filter, g_filter, low1, high1, S_DIM, S_DIM / 2, total4);
    }
    // DWT analysis level 2:  low1 -> low2, high2
    {
        int total4 = (int)(L2 / 4);
        k_dwt_analysis<<<total4 / 256, 256, 0, stream>>>(
            low1, h_filter, g_filter, low2, high2, S_DIM / 2, S_DIM / 4, total4);
    }
    // Linear attention on low2 (in place)
    hipMemsetAsync(kvz, 0, (size_t)B_DIM * HEADS * KVZ_STRIDE * sizeof(float), stream);
    k_attn_kv<<<B_DIM * HEADS * NSEG, 256, 0, stream>>>(low2, kvz);
    k_attn_apply<<<B_DIM * HEADS * (N_LOW / 64), 256, 0, stream>>>(low2, kvz);
    // IWT synthesis level 2: (att=low2, high2) -> rec1
    {
        int total4 = (int)(L1 / 4);
        k_synth<<<total4 / 256, 256, 0, stream>>>(
            low2, high2, h_synth, g_synth, rec1, S_DIM / 4, total4);
    }
    // IWT synthesis level 1 + residual + LayerNorm -> out
    k_synth_ln<<<B_DIM * S_DIM, 256, 0, stream>>>(
        rec1, high1, x, h_synth, g_synth, gamma, beta, out);
}
